// CenterLossN_1099511628281
// MI455X (gfx1250) — matrix operand order repaired and checked by reading, not compiled, not run on hardware
//
#include <hip/hip_runtime.h>
#include <hip/hip_bf16.h>

// ---------------------------------------------------------------------------
// CenterLoss fused kernel for gfx1250 (MI455X, wave32, WMMA).
//
// res[n,c,i,j] = x^2 + centers^2 - 2 * (x[n,c] @ centers[n,c])[i,j]
// loss = sum(clip(max_c softmax_c(res) * label, 1e-12, 1e12)) / (H*W*N)
//
// GEMM path: bf16 operands, f32 accumulate, v_wmma_f32_16x16x32_bf16.
// One wave computes one 16x16 (i,j) tile across all C=11 channels.
// ---------------------------------------------------------------------------

typedef __attribute__((ext_vector_type(16))) __bf16 v16bf;
typedef __attribute__((ext_vector_type(8)))  float  v8f;
typedef __attribute__((ext_vector_type(4)))  float  v4f;

#define N_DIM   32
#define C_DIM   11
#define HW      384
#define TILES   24                // 384 / 16
#define WAVES_PER_BLOCK 4
#define CLAMP_MIN 1e-12f
#define CLAMP_MAX 1e12f

__global__ void zero_out_kernel(float* __restrict__ out) {
    if (threadIdx.x == 0) out[0] = 0.0f;
}

__global__ void __launch_bounds__(WAVES_PER_BLOCK * 32)
center_loss_kernel(const float* __restrict__ x,
                   const float* __restrict__ centers,
                   const int*   __restrict__ labels,
                   float*       __restrict__ out)
{
    const int lane = threadIdx.x & 31;
    const int wib  = threadIdx.x >> 5;

    // Global wave id -> (n, tile_i, tile_j)
    const int w  = blockIdx.x * WAVES_PER_BLOCK + wib;
    const int n  = w / (TILES * TILES);
    const int t  = w % (TILES * TILES);
    const int i0 = (t / TILES) * 16;
    const int j0 = (t % TILES) * 16;

    const int hl  = lane >> 4;      // 0: lanes 0-15, 1: lanes 16-31
    const int l16 = lane & 15;

    const size_t plane = (size_t)HW * HW;

    // A-fragment row for this lane (16x32 bf16 tile of x)
    const int arow = i0 + l16;

    v8f acc[C_DIM];

    #pragma unroll
    for (int c = 0; c < C_DIM; ++c) {
        const float* __restrict__ xa = x       + ((size_t)n * C_DIM + c) * plane;
        const float* __restrict__ cb = centers + ((size_t)n * C_DIM + c) * plane;

        v8f accv{};   // zero f32 accumulator for this channel

        for (int k0 = 0; k0 < HW; k0 += 32) {
            // ---- A fragment: lane<16 holds K {0..7,16..23}, lane>=16 holds
            //      K {8..15,24..31} of row arow (ISA 16-bit A 16x32 layout).
            const float* ap = xa + (size_t)arow * HW + k0 + hl * 8;
            v4f a0 = *(const v4f*)(ap + 0);
            v4f a1 = *(const v4f*)(ap + 4);
            v4f a2 = *(const v4f*)(ap + 16);
            v4f a3 = *(const v4f*)(ap + 20);
            v16bf afrag;
            #pragma unroll
            for (int e = 0; e < 4; ++e) {
                afrag[e]      = (__bf16)a0[e];
                afrag[e + 4]  = (__bf16)a1[e];
                afrag[e + 8]  = (__bf16)a2[e];
                afrag[e + 12] = (__bf16)a3[e];
            }

            // ---- B fragment: lane holds column N=j0+l16, K = hl*8 + tt (tt<8),
            //      16 + hl*8 + (tt-8) (tt>=8), matching A. Loads coalesce across lanes.
            const float* bp = cb + (size_t)(k0 + hl * 8) * HW + (j0 + l16);
            v16bf bfrag;
            #pragma unroll
            for (int tt = 0; tt < 16; ++tt)
                bfrag[tt] = (__bf16)bp[(size_t)(tt + (tt >> 3) * 8) * HW];

            // D = A x B + C  (bf16 in, f32 accumulate)
            accv = __builtin_amdgcn_wmma_f32_16x16x32_bf16(
                false, afrag, false, bfrag, (short)0, accv, false, false);
        }
        acc[c] = accv;
    }

    // ---- Epilogue: res = x^2 + centers^2 - 2*mm ; softmax over C, max,
    //      label weighting, clamp, mean-scaled accumulation.
    float lsum = 0.0f;
    #pragma unroll
    for (int v = 0; v < 8; ++v) {
        const int i = i0 + v + hl * 8;     // C/D layout: M = v + 8*half
        const int j = j0 + l16;            // N = lane & 15
        const size_t pix = (size_t)i * HW + j;

        float r[C_DIM];
        float m = -3.0e38f;
        #pragma unroll
        for (int c = 0; c < C_DIM; ++c) {
            const size_t idx = ((size_t)n * C_DIM + c) * plane + pix;
            const float xv = x[idx];
            const float cv = centers[idx];
            const float rr = xv * xv + cv * cv - 2.0f * acc[c][v];
            r[c] = rr;
            m = fmaxf(m, rr);
        }
        float s = 0.0f;
        #pragma unroll
        for (int c = 0; c < C_DIM; ++c)
            s += __expf(r[c] - m);

        const float pmax = 1.0f / s;       // max_c softmax_c = exp(0)/sum
        const float lab  = (float)labels[(size_t)n * plane + pix];
        float dist = pmax * lab;
        dist = fminf(fmaxf(dist, CLAMP_MIN), CLAMP_MAX);
        lsum += dist;
    }

    // Wave32 reduction, then one atomic per wave (pre-scaled by 1/(H*W*N)).
    #pragma unroll
    for (int off = 16; off > 0; off >>= 1)
        lsum += __shfl_xor(lsum, off, 32);

    if (lane == 0) {
        const float scale = 1.0f / ((float)HW * (float)HW * (float)N_DIM);
        atomicAdd(out, lsum * scale);
    }
}

extern "C" void kernel_launch(void* const* d_in, const int* in_sizes, int n_in,
                              void* d_out, int out_size, void* d_ws, size_t ws_size,
                              hipStream_t stream) {
    (void)in_sizes; (void)n_in; (void)out_size; (void)d_ws; (void)ws_size;

    const float* x       = (const float*)d_in[0];
    const float* centers = (const float*)d_in[1];
    const int*   labels  = (const int*)d_in[2];
    float*       out     = (float*)d_out;

    // d_out is poisoned by the harness; zero the accumulator first.
    zero_out_kernel<<<1, 32, 0, stream>>>(out);

    // 32 n * 24*24 tiles = 18432 waves; 4 waves (128 threads) per block.
    const int total_waves = N_DIM * TILES * TILES;
    const int blocks = total_waves / WAVES_PER_BLOCK;   // 4608
    center_loss_kernel<<<blocks, WAVES_PER_BLOCK * 32, 0, stream>>>(
        x, centers, labels, out);
}
